// Ex4bitLinearV2_24781961298171
// MI455X (gfx1250) — matrix operand order repaired and checked by reading, not compiled, not run on hardware
//
#include <hip/hip_runtime.h>

typedef __attribute__((ext_vector_type(16))) _Float16 v16h;
typedef __attribute__((ext_vector_type(8)))  _Float16 v8h;
typedef __attribute__((ext_vector_type(2)))  _Float16 h2;
typedef __attribute__((ext_vector_type(8)))  float    v8f;

#define KDIM      4096
#define NDIM      11008
#define MDIM      16
#define GROUP     128
#define NSPLIT    8
#define KSLAB     (KDIM / NSPLIT)   // 512 K per block
#define LDS_PITCH (KSLAB + 8)       // halves; pad keeps 16B align, spreads banks
#define NGRP      (KSLAB / GROUP)   // 4 quant groups per slab

// out[m*N+n] = bias[n]  (harness poisons d_out, we own init)
__global__ void gptq_bias_init(const float* __restrict__ bias,
                               float* __restrict__ out) {
    int idx = blockIdx.x * blockDim.x + threadIdx.x;
    if (idx < MDIM * NDIM) {
        out[idx] = bias[idx % NDIM];
    }
}

__global__ __launch_bounds__(256)
void gptq_wmma_gemm(const float* __restrict__ x,
                    const unsigned* __restrict__ qweight,
                    const unsigned* __restrict__ qzeros,
                    const float* __restrict__ scales,
                    float*       __restrict__ out) {
    __shared__ _Float16 lds_x[MDIM * LDS_PITCH];            // 16,640 B
    __shared__ __attribute__((aligned(16))) float lds_rs[NGRP * MDIM]; // rowsums

    const int tid  = threadIdx.x;
    const int wave = tid >> 5;
    const int lane = tid & 31;
    const int m    = lane & 15;      // token row (A) / column-in-tile (B, C)
    const int hi   = lane >> 4;      // lane half selects K sub-chunk

    const int k0 = blockIdx.y * KSLAB;
    const int n  = (blockIdx.x * 8 + wave) * 16 + (lane & 15);

    if (tid < NGRP * MDIM) lds_rs[tid] = 0.0f;
    __syncthreads();

    // ---- Stage x[16, k0:k0+512] into LDS as f16, K-permuted per octet ----
    // lds[oct*8 + jj] = x[oct*8 + pi(jj)], pi = [0,4,1,5,2,6,3,7]: matches the
    // (nibble p, nibble p+4) pair order of the masked dequant. Interleaving two
    // float4s IS that permutation: 2x global b128 -> cvt -> 1x ds_store_b128.
    // Also accumulate f32 row-sums of the *f16-rounded* x per (group, m):
    // needed to cancel the 1024+z1 magic offset analytically after the WMMAs.
    for (int o = tid; o < MDIM * (KSLAB / 8); o += 256) {   // 1024 K-octets
        int row = o >> 6;          // /64 octets per row
        int oct = o & 63;
        const float* px = x + row * KDIM + k0 + oct * 8;    // 16B aligned
        float4 f0 = *(const float4*)(px);
        float4 f1 = *(const float4*)(px + 4);
        union { h2 p[4]; v8h v; } st;
        st.p[0] = (h2){ (_Float16)f0.x, (_Float16)f1.x };   // pos 0,1 = cols 0,4
        st.p[1] = (h2){ (_Float16)f0.y, (_Float16)f1.y };   // pos 2,3 = cols 1,5
        st.p[2] = (h2){ (_Float16)f0.z, (_Float16)f1.z };   // pos 4,5 = cols 2,6
        st.p[3] = (h2){ (_Float16)f0.w, (_Float16)f1.w };   // pos 6,7 = cols 3,7
        *(v8h*)&lds_x[row * LDS_PITCH + oct * 8] = st.v;

        float sum = (((float)st.p[0].x + (float)st.p[0].y)
                   + ((float)st.p[1].x + (float)st.p[1].y))
                  + (((float)st.p[2].x + (float)st.p[2].y)
                   + ((float)st.p[3].x + (float)st.p[3].y));
        atomicAdd(&lds_rs[(oct >> 4) * MDIM + row], sum);   // ds_add_f32
    }
    __syncthreads();

    v8f c = {};

    const int zshift = (n & 7) * 4;
    const int zcol   = n >> 3;

    for (int gi = 0; gi < NGRP; ++gi) {
        const int g = (k0 >> 7) + gi;

        const float    s  = scales[g * NDIM + n];
        const unsigned zw = qzeros[g * (NDIM / 8) + zcol];
        const unsigned z1 = ((zw >> zshift) & 0xFu) + 1u;   // GPTQ zero + 1

        // Unscaled, un-zeroed accumulator: B carries raw f16(1024+q).
        v8f cg = {};

        #pragma unroll
        for (int ks = 0; ks < GROUP / 32; ++ks) {      // 4 x K=32 WMMA steps
            const int kb = gi * GROUP + ks * 32;       // K offset within slab
            const int kg = k0 + kb;                    // global K base

            // ---- A from LDS (already permuted to match B's pair order) ----
            const _Float16* arow = &lds_x[m * LDS_PITCH + kb + hi * 16];
            v8h alo = *(const v8h*)(arow);             // 16B aligned
            v8h ahi = *(const v8h*)(arow + 8);
            v16h a  = __builtin_shufflevector(alo, ahi,
                                              0, 1, 2, 3, 4, 5, 6, 7,
                                              8, 9, 10, 11, 12, 13, 14, 15);

            // ---- B: two packed words per lane; 16 lanes -> 64B coalesced ----
            const int wrow = (kg >> 3) + hi * 2;
            const unsigned w0 = qweight[wrow * NDIM + n];
            const unsigned w1 = qweight[(wrow + 1) * NDIM + n];

            // Pure bit-op dequant: dword p = f16 pair (1024+q_p, 1024+q_{p+4}).
            // No f16 arithmetic in the stream path at all.
            union { v16h v; unsigned u[8]; } bu;
            #pragma unroll
            for (int pI = 0; pI < 4; ++pI) {
                bu.u[pI]     = ((w0 >> (4 * pI)) & 0x000F000Fu) | 0x64006400u;
                bu.u[4 + pI] = ((w1 >> (4 * pI)) & 0x000F000Fu) | 0x64006400u;
            }

            // cg = A x B + cg   (16x16x32 f16 -> f32); first step gets C=0
            cg = __builtin_amdgcn_wmma_f32_16x16x32_f16(
                    /*neg_a=*/false, a, /*neg_b=*/false, bu.v,
                    /*c_mod=*/(short)0, cg, /*reuse_a=*/false, /*reuse_b=*/false);
        }

        // c += s*cg - s*(1024+z1)*rowsum   (exact cancellation of the offset:
        // WMMA summed (1024+q)*x_h; rowsums are over the same f16-rounded x_h)
        const float t = s * (float)(1024u + z1);
        const float* rsg = &lds_rs[gi * MDIM + hi * 8];
        float4 r0 = *(const float4*)(rsg);             // 32B/16B aligned
        float4 r1 = *(const float4*)(rsg + 4);
        const float rs[8] = { r0.x, r0.y, r0.z, r0.w, r1.x, r1.y, r1.z, r1.w };
        #pragma unroll
        for (int i = 0; i < 8; ++i) {
            c[i] = fmaf(s, cg[i], fmaf(-t, rs[i], c[i]));
        }
    }

    // ---- C/D layout: VGPR i, lane L -> M = i + 8*(L>=16), N = L&15 ----
    #pragma unroll
    for (int i = 0; i < 8; ++i) {
        int mm = i + hi * 8;
        atomicAdd(&out[mm * NDIM + n], c[i]);
    }
}

extern "C" void kernel_launch(void* const* d_in, const int* in_sizes, int n_in,
                              void* d_out, int out_size, void* d_ws, size_t ws_size,
                              hipStream_t stream) {
    const float*    x       = (const float*)d_in[0];
    const unsigned* qweight = (const unsigned*)d_in[1];
    const unsigned* qzeros  = (const unsigned*)d_in[2];
    const float*    scales  = (const float*)d_in[3];
    const float*    bias    = (const float*)d_in[4];
    float*          out     = (float*)d_out;

    const int total = MDIM * NDIM;
    gptq_bias_init<<<(total + 255) / 256, 256, 0, stream>>>(bias, out);

    // 86 blocks in N (8 waves x 16 cols each) x 8 K-splits = 688 blocks, 5504 waves
    dim3 grid(NDIM / 128, NSPLIT);
    gptq_wmma_gemm<<<grid, 256, 0, stream>>>(x, qweight, qzeros, scales, out);
}
